// JordanRNN_39857296507579
// MI455X (gfx1250) — matrix operand order repaired and checked by reading, not compiled, not run on hardware
//
#include <hip/hip_runtime.h>
#include <stdint.h>

// ---------------------------------------------------------------------------
// Jordan-feedback GRU, MI455X (gfx1250).
// Serial recurrence -> single persistent 1024-thread workgroup (32 wave32 on
// one WGP). h / y / x_t live in LDS (112KB of 320KB pool). Weights converted
// once fp32->bf16 into d_ws (L2-resident, ~700KB), GEMMs via
// v_wmma_f32_16x16x32_bf16 with fp32 accumulate, gates in fp32 VALU.
// r/z gates accumulate input-side AND hidden-side GEMMs into a single WMMA
// accumulator chain (GRU math needs only the sum there); only the n-gate
// keeps both sides separate. 4 accumulators/task instead of 6 -> no spills.
// ---------------------------------------------------------------------------

typedef __attribute__((ext_vector_type(8)))  __bf16 v8bf;
typedef __attribute__((ext_vector_type(16))) __bf16 v16bf;
typedef __attribute__((ext_vector_type(8)))  float  v8f;

#define T_SEQ 2048
#define B_    128
#define I_    128
#define H_    256
#define O_    64
#define GI_   192   // I_ + O_ (gru input width)

__device__ __forceinline__ unsigned short f2b(float f) {
  unsigned int u = __float_as_uint(f);
  u += 0x7FFFu + ((u >> 16) & 1u);          // round-to-nearest-even
  return (unsigned short)(u >> 16);
}
__device__ __forceinline__ float b2f(unsigned short b) {
  return __uint_as_float(((unsigned int)b) << 16);
}

// A-matrix fragment (16-bit 16x32): lane holds row M=lane&15,
// K = half*16 + [0..15] (same k order as B).
// -> two contiguous 16B loads at +0 and +8 elements.
__device__ __forceinline__ v16bf frag_a(const unsigned short* p) {
  v8bf lo = *(const v8bf*)(p);
  v8bf hi = *(const v8bf*)(p + 8);
  v16bf r;
#pragma unroll
  for (int i = 0; i < 8; ++i) { r[i] = lo[i]; r[i + 8] = hi[i]; }
  return r;
}

// B-matrix fragment (16-bit 32x16): lane holds column N=lane&15,
// K = half*16 + [0..15] -> 16 contiguous bf16 (two 16B loads).
__device__ __forceinline__ v16bf frag_b(const unsigned short* p) {
  v8bf lo = *(const v8bf*)(p);
  v8bf hi = *(const v8bf*)(p + 8);
  v16bf r;
#pragma unroll
  for (int i = 0; i < 8; ++i) { r[i] = lo[i]; r[i + 8] = hi[i]; }
  return r;
}

__device__ __forceinline__ v8f wmma_bf16(v16bf a, v16bf b, v8f c) {
  // (neg_a, A, neg_b, B, c_mod, C, reuse_a, reuse_b)
  return __builtin_amdgcn_wmma_f32_16x16x32_bf16(false, a, false, b,
                                                 (short)0, c, false, false);
}

__device__ __forceinline__ float sigmoid_f(float x) {
  return 1.0f / (1.0f + __expf(-x));
}

// Opaque zero: value is 0 but the compiler cannot prove it, so addresses
// derived from (base + opaque_zero()) are not loop-invariant -> weight loads
// stay inside the t-loop. Base pointer provenance is preserved -> global
// address space + saddr-form global_load_b128.
__device__ __forceinline__ int opaque_zero() {
  int z = 0;
  asm volatile("" : "+s"(z));
  return z;
}

// ---------------------------------------------------------------------------
// Weight conversion: fp32 -> bf16 rows kept row-major (layouts match WMMA
// fragment loads directly, no transpose/packing needed).
// ws layout (ushort units): [0) w_ih 768x192 | 147456) w_hh 768x256 |
//                            344064) fc_w 64x256
// ---------------------------------------------------------------------------
__global__ void cvt_weights(const float* __restrict__ wih,
                            const float* __restrict__ whh,
                            const float* __restrict__ wfc,
                            unsigned short* __restrict__ ws) {
  int i = blockIdx.x * 256 + threadIdx.x;
  if (i < 768 * GI_) ws[i] = f2b(wih[i]);
  if (i < 768 * H_)  ws[768 * GI_ + i] = f2b(whh[i]);
  if (i < O_ * H_)   ws[768 * GI_ + 768 * H_ + i] = f2b(wfc[i]);
}

// ---------------------------------------------------------------------------
// Persistent sequential kernel: one workgroup, 32 waves, loops t=0..2047.
// ---------------------------------------------------------------------------
__global__ __launch_bounds__(1024, 1)
void jordan_gru_seq(const float* __restrict__ x,        // [B][T][I]
                    const float* __restrict__ b_ih,     // [768]
                    const float* __restrict__ b_hh,     // [768]
                    const float* __restrict__ fc_b,     // [64]
                    const unsigned short* __restrict__ wih0, // bf16 [768][192]
                    const unsigned short* __restrict__ whh0, // bf16 [768][256]
                    const unsigned short* __restrict__ wfc0, // bf16 [64][256]
                    float* __restrict__ out)            // [B][O]
{
  __shared__ __align__(16) unsigned short hs[B_ * H_];  // 64 KB  hidden state
  __shared__ __align__(16) unsigned short ys[B_ * O_];  // 16 KB  jordan feedback
  __shared__ __align__(16) unsigned short xs[B_ * I_];  // 32 KB  staged x_t

  const int tid  = threadIdx.x;
  const int wave = tid >> 5;
  const int lane = tid & 31;
  const int hl   = lane >> 4;   // which 16-lane half
  const int ln   = lane & 15;

  // h0 = 0, y0 = 0
#pragma unroll
  for (int k = 0; k < (B_ * H_) / 1024; ++k) hs[tid + k * 1024] = 0;
#pragma unroll
  for (int k = 0; k < (B_ * O_) / 1024; ++k) ys[tid + k * 1024] = 0;

  // Gate-phase assignment: each wave owns gate-column tile nT = wave&15 and
  // 4 batch tiles mT = (wave>>4) + 2*it. Per-lane gate column (invariant):
  const int nT = wave & 15;
  const int jr = nT * 16 + ln;                 // unit index 0..255
  const float bias_r  = b_ih[jr]           + b_hh[jr];
  const float bias_z  = b_ih[H_ + jr]      + b_hh[H_ + jr];
  const float bias_in = b_ih[2 * H_ + jr];
  const float bias_hn = b_hh[2 * H_ + jr];

  // fc-phase assignment (one 16x16 tile per wave):
  const int fmT = wave >> 2;
  const int foT = wave & 3;
  const int fo  = foT * 16 + ln;
  const float fbias = fc_b[fo];

  __syncthreads();

  for (int t = 0; t < T_SEQ; ++t) {
    // Per-iteration opaque zero offset: defeats LICM of weight loads while
    // keeping global addrspace + scalar-base addressing.
    const int zz = opaque_zero();
    const unsigned short* wih = wih0 + zz;
    const unsigned short* whh = whh0 + zz;
    const unsigned short* wfc = wfc0 + zz;

    // ---- stage x_t into LDS as bf16 (x read exactly once overall) ----
#pragma unroll
    for (int k = 0; k < (B_ * I_) / 1024; ++k) {
      int idx = tid + k * 1024;
      int b = idx >> 7, i = idx & 127;
      size_t g = (size_t)b * (T_SEQ * I_) + (size_t)t * I_ + i;
      xs[idx] = f2b(x[g]);
      if (t + 1 < T_SEQ) __builtin_prefetch(&x[g + I_], 0, 1);
    }
    __syncthreads();

    // ---- phase 1: gates + h_new (kept in registers, deferred writeback) ----
    float hnew[4][8];
#pragma unroll
    for (int it = 0; it < 4; ++it) {
      const int mT  = (wave + it * 32) >> 4;
      const int row = mT * 16 + ln;
      // r/z: gi and gh summed in one WMMA accumulator chain each.
      // n : input side (accIN) and hidden side (accHN) kept separate.
      v8f accR = {}, accZ = {}, accIN = {}, accHN = {};

      // gi, x-part: K = 128
#pragma unroll
      for (int kT = 0; kT < 4; ++kT) {
        v16bf a  = frag_a(&xs[row * I_ + kT * 32 + hl * 16]);
        accR  = wmma_bf16(a, frag_b(&wih[(jr)          * GI_ + kT * 32 + hl * 16]), accR);
        accZ  = wmma_bf16(a, frag_b(&wih[(H_ + jr)     * GI_ + kT * 32 + hl * 16]), accZ);
        accIN = wmma_bf16(a, frag_b(&wih[(2 * H_ + jr) * GI_ + kT * 32 + hl * 16]), accIN);
      }
      // gi, y-feedback part: K = 64 (w_ih columns 128..191)
#pragma unroll
      for (int kT = 0; kT < 2; ++kT) {
        v16bf a  = frag_a(&ys[row * O_ + kT * 32 + hl * 16]);
        accR  = wmma_bf16(a, frag_b(&wih[(jr)          * GI_ + I_ + kT * 32 + hl * 16]), accR);
        accZ  = wmma_bf16(a, frag_b(&wih[(H_ + jr)     * GI_ + I_ + kT * 32 + hl * 16]), accZ);
        accIN = wmma_bf16(a, frag_b(&wih[(2 * H_ + jr) * GI_ + I_ + kT * 32 + hl * 16]), accIN);
      }
      // gh: K = 256 (r/z continue the same accumulators; n-hidden separate)
#pragma unroll
      for (int kT = 0; kT < 8; ++kT) {
        v16bf a  = frag_a(&hs[row * H_ + kT * 32 + hl * 16]);
        accR  = wmma_bf16(a, frag_b(&whh[(jr)          * H_ + kT * 32 + hl * 16]), accR);
        accZ  = wmma_bf16(a, frag_b(&whh[(H_ + jr)     * H_ + kT * 32 + hl * 16]), accZ);
        accHN = wmma_bf16(a, frag_b(&whh[(2 * H_ + jr) * H_ + kT * 32 + hl * 16]), accHN);
      }
      // elementwise gates; D layout: VGPR g -> M = g + 8*hl, N = ln
#pragma unroll
      for (int g = 0; g < 8; ++g) {
        int m = mT * 16 + hl * 8 + g;
        float r = sigmoid_f(accR[g] + bias_r);
        float z = sigmoid_f(accZ[g] + bias_z);
        float n = tanhf(accIN[g] + bias_in + r * (accHN[g] + bias_hn));
        float hv = b2f(hs[m * H_ + jr]);
        hnew[it][g] = (1.0f - z) * n + z * hv;
      }
    }
    __syncthreads();   // all reads of h done -> safe to overwrite
#pragma unroll
    for (int it = 0; it < 4; ++it) {
      const int mT = (wave + it * 32) >> 4;
#pragma unroll
      for (int g = 0; g < 8; ++g) {
        int m = mT * 16 + hl * 8 + g;
        hs[m * H_ + jr] = f2b(hnew[it][g]);
      }
    }
    __syncthreads();

    // ---- phase 2: y = tanh(h_new @ fc_w^T + fc_b), one tile per wave ----
    {
      const int row = fmT * 16 + ln;
      v8f acc = {};
#pragma unroll
      for (int kT = 0; kT < 8; ++kT) {
        v16bf a = frag_a(&hs[row * H_ + kT * 32 + hl * 16]);
        v16bf b = frag_b(&wfc[fo * H_ + kT * 32 + hl * 16]);
        acc = wmma_bf16(a, b, acc);
      }
#pragma unroll
      for (int g = 0; g < 8; ++g) {
        int m = fmT * 16 + hl * 8 + g;
        float yv = tanhf(acc[g] + fbias);
        ys[m * O_ + fo] = f2b(yv);
        if (t == T_SEQ - 1) out[m * O_ + fo] = yv;   // final output only
      }
    }
    __syncthreads();
  }
}

// ---------------------------------------------------------------------------
extern "C" void kernel_launch(void* const* d_in, const int* in_sizes, int n_in,
                              void* d_out, int out_size, void* d_ws, size_t ws_size,
                              hipStream_t stream) {
  const float* x   = (const float*)d_in[0];
  const float* wih = (const float*)d_in[1];
  const float* whh = (const float*)d_in[2];
  const float* bih = (const float*)d_in[3];
  const float* bhh = (const float*)d_in[4];
  const float* wfc = (const float*)d_in[5];
  const float* fcb = (const float*)d_in[6];

  unsigned short* ws    = (unsigned short*)d_ws;
  unsigned short* wih_b = ws;
  unsigned short* whh_b = ws + 768 * GI_;
  unsigned short* wfc_b = ws + 768 * GI_ + 768 * H_;

  // one-time fp32 -> bf16 weight conversion (deterministic, re-run each call)
  cvt_weights<<<768, 256, 0, stream>>>(wih, whh, wfc, ws);

  // persistent sequential recurrence: 1 workgroup, 32 wave32
  jordan_gru_seq<<<1, 1024, 0, stream>>>(x, bih, bhh, fcb,
                                         wih_b, whh_b, wfc_b, (float*)d_out);
}
